// MultiHeadAttentionBlock_41283225649156
// MI455X (gfx1250) — matrix operand order repaired and checked by reading, not compiled, not run on hardware
//
#include <hip/hip_runtime.h>
#include <hip/hip_bf16.h>
#include <stdint.h>

// ---------------------------------------------------------------------------
// MI455X (gfx1250) fused MHA block:  bf16 WMMA GEMMs + flash attention.
// wave32, 256 threads (8 waves) per block.
// ---------------------------------------------------------------------------

typedef __attribute__((ext_vector_type(16))) __bf16         v16bf;
typedef __attribute__((ext_vector_type(8)))  float          v8f;
typedef __attribute__((ext_vector_type(8)))  unsigned short v8u16;

static constexpr int D_MODEL = 1024;
static constexpr int HEADS   = 16;
static constexpr int D_K     = 64;
static constexpr int BATCH   = 2;
static constexpr int SEQ     = 2048;
static constexpr int MROWS   = BATCH * SEQ;     // 4096

__device__ __forceinline__ unsigned short f2bf(float f) {
    unsigned int u = __float_as_uint(f);
    u += 0x7FFFu + ((u >> 16) & 1u);            // round-to-nearest-even
    return (unsigned short)(u >> 16);
}
__device__ __forceinline__ float bf2f(unsigned short u) {
    return __uint_as_float(((unsigned int)u) << 16);
}

struct FragBF { union { v16bf bf; v8u16 h[2]; }; };

__device__ __forceinline__ v8f wmma_bf16(v16bf a, v16bf b, v8f c) {
    // D(16x16,f32) = A(16x32,bf16) * B(32x16,bf16) + C
    return __builtin_amdgcn_wmma_f32_16x16x32_bf16(
        /*neg_a=*/false, a, /*neg_b=*/false, b,
        /*c_mod=*/(short)0, c, /*reuse_a=*/false, /*reuse_b=*/false);
}

// ---- xor-shuffle within 16-lane half-waves via v_permlane16_b32 (VALU) ----
constexpr unsigned permsel(int base, int x) {
    unsigned w = 0;
    for (int i = 0; i < 8; ++i)
        w |= (unsigned)(((base + i) ^ x) & 15) << (4 * i);
    return w;
}
template <int X>
__device__ __forceinline__ float permxor16(float v) {
#if __has_builtin(__builtin_amdgcn_permlane16)
    constexpr unsigned lo = permsel(0, X), hi = permsel(8, X);
    unsigned u = __float_as_uint(v);
    u = __builtin_amdgcn_permlane16(u, u, lo, hi, false, false);
    return __uint_as_float(u);
#else
    return __shfl_xor(v, X, 32);
#endif
}
__device__ __forceinline__ float rowmax16(float x) {
    x = fmaxf(x, permxor16<1>(x));
    x = fmaxf(x, permxor16<2>(x));
    x = fmaxf(x, permxor16<4>(x));
    x = fmaxf(x, permxor16<8>(x));
    return x;
}

// ---------------------------------------------------------------------------
// fp32 -> bf16 bulk convert (activations)
// ---------------------------------------------------------------------------
__global__ __launch_bounds__(256)
void cvt_f32_bf16_kernel(const float* __restrict__ src,
                         unsigned short* __restrict__ dst, int n4) {
    int i      = blockIdx.x * blockDim.x + threadIdx.x;
    int stride = gridDim.x * blockDim.x;
    for (; i < n4; i += stride) {
        float4 f = ((const float4*)src)[i];
        unsigned int lo = (unsigned int)f2bf(f.x) | ((unsigned int)f2bf(f.y) << 16);
        unsigned int hi = (unsigned int)f2bf(f.z) | ((unsigned int)f2bf(f.w) << 16);
        ((uint2*)dst)[i] = make_uint2(lo, hi);
    }
}

// ---------------------------------------------------------------------------
// W[N][K] fp32  ->  Wt[K][N] bf16   (so GEMM computes x @ W^T with contiguous B)
// ---------------------------------------------------------------------------
__global__ __launch_bounds__(256)
void wtrans_kernel(const float* __restrict__ W, unsigned short* __restrict__ Wt) {
    __shared__ float t[32][33];
    const int n0 = blockIdx.x * 32, k0 = blockIdx.y * 32;
    for (int i = threadIdx.x; i < 1024; i += 256) {
        int r = i >> 5, c = i & 31;
        t[r][c] = W[(size_t)(n0 + r) * D_MODEL + k0 + c];
    }
    __syncthreads();
    for (int i = threadIdx.x; i < 1024; i += 256) {
        int r = i >> 5, c = i & 31;
        Wt[(size_t)(k0 + r) * D_MODEL + n0 + c] = f2bf(t[c][r]);
    }
}

// ---------------------------------------------------------------------------
// GEMM: C[M=4096][N=1024] = A_bf16[M][1024] * Bt_bf16[1024][N] + bias
// Block tile 128x128, K-step 32, register-staged double-buffered LDS.
// 8 waves as 2(m) x 4(n); wave computes a 64x32 patch (8 WMMA tiles / K-step).
// MODE 0: fp32 flat [m][n]
// MODE 1: bf16 head-split [b,h,s,d]
// MODE 2: bf16 transposed  [b,h,d,s]   (packed b128 stores; K -> QK^T operand)
// ---------------------------------------------------------------------------
template <int MODE>
__global__ __launch_bounds__(256)
void gemm_bf16_kernel(const unsigned short* __restrict__ A,
                      const unsigned short* __restrict__ Bt,
                      const float* __restrict__ bias,
                      void* __restrict__ outp) {
    constexpr int Kd = 1024, Nd = 1024;
    constexpr int ALD = 40;            // 32 + 8 pad
    constexpr int BLD = 136;           // 128 + 8 pad
    constexpr int ASZ = 128 * ALD;
    constexpr int BSZ = 32 * BLD;
    __shared__ unsigned short a_lds[2 * ASZ];
    __shared__ unsigned short b_lds[2 * BSZ];

    const int tid  = threadIdx.x;
    const int lane = tid & 31, wv = tid >> 5;
    const int hf   = lane >> 4, l16 = lane & 15;
    const int wm   = wv >> 2,  wn  = wv & 3;
    const int m0   = blockIdx.y * 128, n0 = blockIdx.x * 128;

    // per-thread staging coordinates (2 x 16B vectors for A and for B)
    const int ar = tid >> 2, ac = (tid & 3) * 8;   // A rows ar, ar+64
    const int br = tid >> 4, bc = (tid & 15) * 8;  // B rows br, br+16

    v8u16 aR0, aR1, bR0, bR1;
    auto load_g = [&](int kb) {
        aR0 = *(const v8u16*)(A + (size_t)(m0 + ar) * Kd + kb + ac);
        aR1 = *(const v8u16*)(A + (size_t)(m0 + ar + 64) * Kd + kb + ac);
        bR0 = *(const v8u16*)(Bt + (size_t)(kb + br) * Nd + n0 + bc);
        bR1 = *(const v8u16*)(Bt + (size_t)(kb + br + 16) * Nd + n0 + bc);
    };
    auto store_l = [&](int buf) {
        *(v8u16*)&a_lds[buf * ASZ + ar * ALD + ac]        = aR0;
        *(v8u16*)&a_lds[buf * ASZ + (ar + 64) * ALD + ac] = aR1;
        *(v8u16*)&b_lds[buf * BSZ + br * BLD + bc]        = bR0;
        *(v8u16*)&b_lds[buf * BSZ + (br + 16) * BLD + bc] = bR1;
    };

    const v8f vzero = {0.f, 0.f, 0.f, 0.f, 0.f, 0.f, 0.f, 0.f};
    v8f acc[4][2];
#pragma unroll
    for (int i = 0; i < 4; ++i)
#pragma unroll
        for (int j = 0; j < 2; ++j) acc[i][j] = vzero;

    load_g(0);
    store_l(0);
    int cur = 0;
    for (int kb = 0; kb < Kd; kb += 32) {
        const bool has_next = (kb + 32 < Kd);
        if (has_next) {
            load_g(kb + 32);                       // overlap with WMMAs below
            if (kb + 64 < Kd) {                    // warm L2 two tiles ahead
                __builtin_prefetch(A + (size_t)(m0 + ar) * Kd + kb + 64, 0, 1);
                __builtin_prefetch(Bt + (size_t)(kb + 64 + br) * Nd + n0 + bc, 0, 1);
            }
        }
        __syncthreads();                           // LDS[cur] ready

        const unsigned short* abase = &a_lds[cur * ASZ];
        const unsigned short* bbase = &b_lds[cur * BSZ];
        FragBF af[4], bfr[2];
#pragma unroll
        for (int i = 0; i < 4; ++i) {
            const unsigned short* p = abase + (wm * 64 + i * 16 + l16) * ALD;
            af[i].h[0] = *(const v8u16*)(p + hf * 8);
            af[i].h[1] = *(const v8u16*)(p + 16 + hf * 8);
        }
#pragma unroll
        for (int j = 0; j < 2; ++j) {
            const unsigned short* p = bbase + (hf * 8) * BLD + wn * 32 + j * 16 + l16;
#pragma unroll
            for (int e = 0; e < 8; ++e) {
                bfr[j].h[0][e] = p[e * BLD];
                bfr[j].h[1][e] = p[(16 + e) * BLD];
            }
        }
#pragma unroll
        for (int i = 0; i < 4; ++i)
#pragma unroll
            for (int j = 0; j < 2; ++j)
                acc[i][j] = wmma_bf16(af[i].bf, bfr[j].bf, acc[i][j]);

        if (has_next) store_l(cur ^ 1);            // fill other buffer
        cur ^= 1;
    }

    // epilogue: D layout -> lane(hf,l16): col n = l16, rows m = vv + 8*hf
#pragma unroll
    for (int i = 0; i < 4; ++i) {
#pragma unroll
        for (int j = 0; j < 2; ++j) {
            const int n  = n0 + wn * 32 + j * 16 + l16;
            const float bn = bias[n];
            const int mb = m0 + wm * 64 + i * 16 + hf * 8;
            if constexpr (MODE == 0) {
#pragma unroll
                for (int vv = 0; vv < 8; ++vv)
                    ((float*)outp)[(size_t)(mb + vv) * Nd + n] = acc[i][j][vv] + bn;
            } else if constexpr (MODE == 1) {
                unsigned short* o = (unsigned short*)outp;
#pragma unroll
                for (int vv = 0; vv < 8; ++vv) {
                    const int m  = mb + vv;
                    const int bb = m >> 11, ss = m & (SEQ - 1);
                    const int hh = n >> 6,  dd = n & (D_K - 1);
                    o[(((size_t)bb * HEADS + hh) * SEQ + ss) * D_K + dd] = f2bf(acc[i][j][vv] + bn);
                }
            } else {  // MODE 2: rows contiguous in s -> one b128 store
                unsigned short* o = (unsigned short*)outp;
                const int bb = mb >> 11, ssb = mb & (SEQ - 1);
                const int hh = n >> 6,   dd  = n & (D_K - 1);
                v8u16 pack;
#pragma unroll
                for (int vv = 0; vv < 8; ++vv) pack[vv] = f2bf(acc[i][j][vv] + bn);
                *(v8u16*)&o[(((size_t)bb * HEADS + hh) * D_K + dd) * SEQ + ssb] = pack;
            }
        }
    }
}

// ---------------------------------------------------------------------------
// Flash attention (causal).  Grid: (SEQ/128, BATCH*HEADS).
// Per block: 128 queries, 8 waves x 16 query rows.  KV blocks of 64,
// register-staged double-buffered LDS.  Q pre-scaled by 1/sqrt(64)=2^-3.
// Row max: permlane16 xor-reduce.  Row sum: P x ones via WMMA.
// ---------------------------------------------------------------------------
__global__ __launch_bounds__(256)
void attn_kernel(const unsigned short* __restrict__ Qh,
                 const unsigned short* __restrict__ Kt,   // [b,h,d,s]
                 const unsigned short* __restrict__ Vh,   // [b,h,s,d]
                 unsigned short* __restrict__ ctx) {      // [b,s,h*64+d]
    constexpr int LD  = 72;            // 64 + 8 pad
    constexpr int KSZ = 64 * LD;
    __shared__ unsigned short q_lds[128 * LD];
    __shared__ unsigned short kt_lds[2 * KSZ];
    __shared__ unsigned short v_lds[2 * KSZ];
    __shared__ unsigned short p_lds[8 * 16 * LD];

    const int tid  = threadIdx.x;
    const int lane = tid & 31, wv = tid >> 5;
    const int hf   = lane >> 4, l16 = lane & 15;
    const int q0   = blockIdx.x * 128;
    const int bh   = blockIdx.y;
    const int b    = bh >> 4, h = bh & 15;

    const unsigned short* Qb  = Qh + (size_t)bh * SEQ * D_K;
    const unsigned short* Ktb = Kt + (size_t)bh * D_K * SEQ;
    const unsigned short* Vb  = Vh + (size_t)bh * SEQ * D_K;

    // ---- stage Q tile (128x64), pre-scale by 0.125 (exact in bf16) ----
    for (int it = tid; it < 128 * 64 / 8; it += 256) {
        int r = it >> 3, c8 = (it & 7) * 8;
        v8u16 in = *(const v8u16*)(Qb + (size_t)(q0 + r) * D_K + c8);
        v8u16 ov;
#pragma unroll
        for (int e = 0; e < 8; ++e) ov[e] = f2bf(bf2f(in[e]) * 0.125f);
        *(v8u16*)&q_lds[r * LD + c8] = ov;
    }

    // KV staging: each thread owns rows r0, r0+32 (16B vectors)
    const int r0 = tid >> 3, c8 = (tid & 7) * 8;
    v8u16 kR0, kR1, vR0, vR1;
    auto load_kv = [&](int kv0) {
        kR0 = *(const v8u16*)(Ktb + (size_t)r0 * SEQ + kv0 + c8);
        kR1 = *(const v8u16*)(Ktb + (size_t)(r0 + 32) * SEQ + kv0 + c8);
        vR0 = *(const v8u16*)(Vb + (size_t)(kv0 + r0) * D_K + c8);
        vR1 = *(const v8u16*)(Vb + (size_t)(kv0 + r0 + 32) * D_K + c8);
    };
    auto store_kv = [&](int buf) {
        *(v8u16*)&kt_lds[buf * KSZ + r0 * LD + c8]        = kR0;
        *(v8u16*)&kt_lds[buf * KSZ + (r0 + 32) * LD + c8] = kR1;
        *(v8u16*)&v_lds[buf * KSZ + r0 * LD + c8]         = vR0;
        *(v8u16*)&v_lds[buf * KSZ + (r0 + 32) * LD + c8]  = vR1;
    };
    load_kv(0);
    store_kv(0);
    __syncthreads();

    FragBF qf[2];
#pragma unroll
    for (int kc = 0; kc < 2; ++kc) {
        const unsigned short* p = &q_lds[(wv * 16 + l16) * LD + kc * 32];
        qf[kc].h[0] = *(const v8u16*)(p + hf * 8);
        qf[kc].h[1] = *(const v8u16*)(p + 16 + hf * 8);
    }
    FragBF ones;                                    // bf16 1.0 everywhere
#pragma unroll
    for (int e = 0; e < 8; ++e) { ones.h[0][e] = 0x3F80; ones.h[1][e] = 0x3F80; }

    const v8f vzero = {0.f, 0.f, 0.f, 0.f, 0.f, 0.f, 0.f, 0.f};
    v8f acc[4];
#pragma unroll
    for (int jd = 0; jd < 4; ++jd) acc[jd] = vzero;
    float m_run[8], l_run[8];
#pragma unroll
    for (int vv = 0; vv < 8; ++vv) { m_run[vv] = -3.0e38f; l_run[vv] = 0.f; }

    const int nkv = q0 + 128;  // causal upper bound for this query block
    int cur = 0;
    for (int kv0 = 0; kv0 < nkv; kv0 += 64) {
        const bool has_next = (kv0 + 64 < nkv);
        if (has_next) load_kv(kv0 + 64);            // overlap with compute
        __syncthreads();                            // LDS[cur] ready

        const unsigned short* ktb = &kt_lds[cur * KSZ];
        const unsigned short* vtb = &v_lds[cur * KSZ];

        // ---- scores: 4 tiles of 16 keys, each K=64 via 2 WMMAs ----
        v8f sc[4];
#pragma unroll
        for (int j = 0; j < 4; ++j) {
            v8f c = vzero;
#pragma unroll
            for (int kc = 0; kc < 2; ++kc) {
                FragBF bfK;
                const unsigned short* p = ktb + (kc * 32 + hf * 8) * LD + j * 16 + l16;
#pragma unroll
                for (int e = 0; e < 8; ++e) {
                    bfK.h[0][e] = p[e * LD];
                    bfK.h[1][e] = p[(16 + e) * LD];
                }
                c = wmma_bf16(qf[kc].bf, bfK.bf, c);
            }
            sc[j] = c;
        }

        // ---- causal mask + online softmax (rows split across half-waves) ----
        float scl[8];
#pragma unroll
        for (int vv = 0; vv < 8; ++vv) {
            const int qidx = q0 + wv * 16 + hf * 8 + vv;
            float mloc = -3.0e38f;
#pragma unroll
            for (int j = 0; j < 4; ++j) {
                const int kidx = kv0 + j * 16 + l16;
                float s = sc[j][vv];
                if (kidx > qidx) s = -1.0e30f;
                sc[j][vv] = s;
                mloc = fmaxf(mloc, s);
            }
            mloc = rowmax16(mloc);                  // permlane16 xor-reduce
            const float mn = fmaxf(m_run[vv], mloc);
            scl[vv] = __expf(m_run[vv] - mn);
            m_run[vv] = mn;
#pragma unroll
            for (int j = 0; j < 4; ++j) sc[j][vv] = __expf(sc[j][vv] - mn);
#pragma unroll
            for (int jd = 0; jd < 4; ++jd) acc[jd][vv] *= scl[vv];
        }

        // ---- P: D-layout regs -> per-wave LDS -> A-fragment layout ----
        unsigned short* pw = &p_lds[wv * 16 * LD];
#pragma unroll
        for (int j = 0; j < 4; ++j)
#pragma unroll
            for (int vv = 0; vv < 8; ++vv)
                pw[(hf * 8 + vv) * LD + j * 16 + l16] = f2bf(sc[j][vv]);
        asm volatile("s_wait_dscnt 0" ::: "memory");  // wave-local LDS RAW

        FragBF pf[2];
#pragma unroll
        for (int kc = 0; kc < 2; ++kc) {
            const unsigned short* p = &pw[l16 * LD + kc * 32];
            pf[kc].h[0] = *(const v8u16*)(p + hf * 8);
            pf[kc].h[1] = *(const v8u16*)(p + 16 + hf * 8);
        }

        // ---- row sums on the matrix unit: rsum = P x ones ----
        v8f ssum = wmma_bf16(pf[0].bf, ones.bf, vzero);
        ssum     = wmma_bf16(pf[1].bf, ones.bf, ssum);
#pragma unroll
        for (int vv = 0; vv < 8; ++vv)
            l_run[vv] = l_run[vv] * scl[vv] + ssum[vv];

        // ---- PV: acc(16x64) += P(16x64) * V(64x64) ----
#pragma unroll
        for (int jd = 0; jd < 4; ++jd) {
#pragma unroll
            for (int kc = 0; kc < 2; ++kc) {
                FragBF vf;
                const unsigned short* p = vtb + (kc * 32 + hf * 8) * LD + jd * 16 + l16;
#pragma unroll
                for (int e = 0; e < 8; ++e) {
                    vf.h[0][e] = p[e * LD];
                    vf.h[1][e] = p[(16 + e) * LD];
                }
                acc[jd] = wmma_bf16(pf[kc].bf, vf.bf, acc[jd]);
            }
        }

        if (has_next) store_kv(cur ^ 1);
        cur ^= 1;
    }

    // ---- normalize + merged-head store (bf16) ----
#pragma unroll
    for (int jd = 0; jd < 4; ++jd) {
#pragma unroll
        for (int vv = 0; vv < 8; ++vv) {
            const int s = q0 + wv * 16 + hf * 8 + vv;
            const float val = acc[jd][vv] / l_run[vv];
            ctx[((size_t)b * SEQ + s) * D_MODEL + h * D_K + jd * 16 + l16] = f2bf(val);
        }
    }
}

// ---------------------------------------------------------------------------
// Launch: prep (convert/transpose) -> QKV GEMMs -> flash attention -> O GEMM
// Workspace layout (bytes), total 64 MB:
//   0:qb 8M | 8M:kb | 16M:vb | 24M..30M: Wqt/Wkt/Wvt/Wot (2M each)
//   32M:Qh | 40M:Kt | 48M:Vh | 56M:ctx
// ---------------------------------------------------------------------------
extern "C" void kernel_launch(void* const* d_in, const int* in_sizes, int n_in,
                              void* d_out, int out_size, void* d_ws, size_t ws_size,
                              hipStream_t stream) {
    (void)in_sizes; (void)n_in; (void)out_size; (void)ws_size;

    const float* q  = (const float*)d_in[0];
    const float* k  = (const float*)d_in[1];
    const float* v  = (const float*)d_in[2];
    // d_in[3] = causal mask (implemented analytically)
    const float* Wq = (const float*)d_in[4];  const float* bq = (const float*)d_in[5];
    const float* Wk = (const float*)d_in[6];  const float* bk = (const float*)d_in[7];
    const float* Wv = (const float*)d_in[8];  const float* bv = (const float*)d_in[9];
    const float* Wo = (const float*)d_in[10]; const float* bo = (const float*)d_in[11];
    float* out = (float*)d_out;

    char* ws = (char*)d_ws;
    const size_t MB = 1ull << 20;
    unsigned short* qb  = (unsigned short*)(ws + 0 * MB);
    unsigned short* kb  = (unsigned short*)(ws + 8 * MB);
    unsigned short* vb  = (unsigned short*)(ws + 16 * MB);
    unsigned short* Wqt = (unsigned short*)(ws + 24 * MB);
    unsigned short* Wkt = (unsigned short*)(ws + 26 * MB);
    unsigned short* Wvt = (unsigned short*)(ws + 28 * MB);
    unsigned short* Wot = (unsigned short*)(ws + 30 * MB);
    unsigned short* Qh  = (unsigned short*)(ws + 32 * MB);
    unsigned short* Kt  = (unsigned short*)(ws + 40 * MB);
    unsigned short* Vh  = (unsigned short*)(ws + 48 * MB);
    unsigned short* ctx = (unsigned short*)(ws + 56 * MB);

    const int n4 = MROWS * D_MODEL / 4;  // 1M float4s
    cvt_f32_bf16_kernel<<<1024, 256, 0, stream>>>(q, qb, n4);
    cvt_f32_bf16_kernel<<<1024, 256, 0, stream>>>(k, kb, n4);
    cvt_f32_bf16_kernel<<<1024, 256, 0, stream>>>(v, vb, n4);

    dim3 tg(32, 32);
    wtrans_kernel<<<tg, 256, 0, stream>>>(Wq, Wqt);
    wtrans_kernel<<<tg, 256, 0, stream>>>(Wk, Wkt);
    wtrans_kernel<<<tg, 256, 0, stream>>>(Wv, Wvt);
    wtrans_kernel<<<tg, 256, 0, stream>>>(Wo, Wot);

    dim3 gg(D_MODEL / 128, MROWS / 128);  // (8, 32)
    gemm_bf16_kernel<1><<<gg, 256, 0, stream>>>(qb, Wqt, bq, Qh);   // Q: [b,h,s,d]
    gemm_bf16_kernel<2><<<gg, 256, 0, stream>>>(kb, Wkt, bk, Kt);   // K: [b,h,d,s]
    gemm_bf16_kernel<1><<<gg, 256, 0, stream>>>(vb, Wvt, bv, Vh);   // V: [b,h,s,d]

    attn_kernel<<<dim3(SEQ / 128, BATCH * HEADS), 256, 0, stream>>>(Qh, Kt, Vh, ctx);

    gemm_bf16_kernel<0><<<gg, 256, 0, stream>>>(ctx, Wot, bo, out); // fp32 out
}
